// DA_HGAN_1614907703512
// MI455X (gfx1250) — matrix operand order repaired and checked by reading, not compiled, not run on hardware
//
#include <hip/hip_runtime.h>
#include <math.h>

// DA-HGAN forward for MI455X (gfx1250).
// N=8192 nodes, M=2048 hyperedges, D=256, DH=8. H is 67MB -> fits in 192MB L2,
// so multiple row-major passes cost ~one HBM read. Density/rho terms cancel in
// the masked softmax (per-row constants; masked entries exp(-1e12)==0).

#define N_NODES 8192
#define M_EDGES 2048
#define D_IN    256
#define DH      8
#define LEAKSL  0.2f

typedef __attribute__((ext_vector_type(16))) _Float16 v16h;
typedef __attribute__((ext_vector_type(8)))  float    v8f;
typedef __attribute__((ext_vector_type(2)))  _Float16 h2;

__device__ __forceinline__ float leaky(float x) { return x > 0.0f ? x : LEAKSL * x; }
__device__ __forceinline__ float elu1(float x)  { return x > 0.0f ? x : expm1f(x); }

__device__ __forceinline__ unsigned pack_h2(float x, float y) {
    h2 p; p.x = (_Float16)x; p.y = (_Float16)y;
    return *(unsigned*)&p;
}

// order-preserving float<->uint encoding for atomicMax on floats (any sign)
__device__ __forceinline__ unsigned f2o(float f) {
    unsigned u = __float_as_uint(f);
    return (u & 0x80000000u) ? ~u : (u | 0x80000000u);
}
__device__ __forceinline__ float o2f(unsigned u) {
    return (u & 0x80000000u) ? __uint_as_float(u & 0x7fffffffu) : __uint_as_float(~u);
}

// ---------------------------------------------------------------------------
// K0: zero the accumulators (d_ws is poisoned with 0xAA; must re-init per call)
// ---------------------------------------------------------------------------
__global__ void init_kernel(float* __restrict__ E_acc, unsigned* __restrict__ t_col_u,
                            float* __restrict__ s_col, float* __restrict__ U_e,
                            float* __restrict__ U_x) {
    int i = blockIdx.x * blockDim.x + threadIdx.x;
    if (i < M_EDGES * DH) E_acc[i] = 0.0f;
    if (i < M_EDGES) { t_col_u[i] = 0u; s_col[i] = 0.0f; }  // 0u == "empty column" sentinel
    if (i < DH) { U_e[i] = 0.0f; U_x[i] = 0.0f; }
}

// ---------------------------------------------------------------------------
// K1: P[rows,8] = A[rows,256] @ W[256,8] via v_wmma_f32_16x16x32_f16.
// One wave per 16-row tile; K=256 -> 8 WMMA steps. A is staged in LDS as f16;
// W is staged transposed + zero-padded to 16 columns in f16, so every WMMA
// operand pair is a single unconditional ds_load_b32 (branchless).
// ---------------------------------------------------------------------------
__global__ void proj_wmma(const float* __restrict__ A, const float* __restrict__ W,
                          float* __restrict__ P, int rows) {
    __shared__ _Float16 tAh[16 * D_IN];   // 8KB: A tile, row-major [16][256]
    __shared__ _Float16 tWh[16 * D_IN];   // 8KB: W^T padded, [col(16)][K(256)]
    unsigned* tAu = (unsigned*)tAh;
    unsigned* tWu = (unsigned*)tWh;
    const int lane = threadIdx.x;         // blockDim.x == 32
    const int row0 = blockIdx.x * 16;

    // zero W^T pad (2048 dwords)
#pragma unroll
    for (int j = 0; j < 64; ++j) tWu[lane + 32 * j] = 0u;
    __syncthreads();

    // stage A tile as packed f16 (coalesced float4 global reads)
    const float4* A4 = (const float4*)(A + (size_t)row0 * D_IN);
#pragma unroll
    for (int j = 0; j < 32; ++j) {
        int f = lane + 32 * j;            // float4 index within tile
        float4 v = A4[f];
        tAu[2 * f]     = pack_h2(v.x, v.y);
        tAu[2 * f + 1] = pack_h2(v.z, v.w);
    }
    // stage W transposed: tWh[col*256 + K] = W[K*8 + col]
#pragma unroll
    for (int j = 0; j < 64; ++j) {
        int idx = lane + 32 * j;          // 0..2047
        int K = idx >> 3, col = idx & 7;
        tWh[col * D_IN + K] = (_Float16)W[idx];
    }
    __syncthreads();

    const int m    = lane & 15;           // A row within tile / B column
    const int half = lane >> 4;
    v8f c = {};
#pragma unroll
    for (int kk = 0; kk < 8; ++kk) {
        union { v16h v; unsigned u[8]; } a, b;
#pragma unroll
        for (int jj = 0; jj < 8; ++jj) {
            // A k order used here (contiguous): lanes0-15 K=0..15 ; lanes16-31 K=16..31
            int Kp = kk * 32 + 16 * half + 2 * jj;
            a.u[jj] = tAu[(m * D_IN + Kp) >> 1];
            // B k order used here (same as A): col = lane&15; lanes0-15 K=0..15, lanes16-31 K=16..31
            int Kb = kk * 32 + 16 * half + 2 * jj;
            b.u[jj] = tWu[(m * D_IN + Kb) >> 1];
        }
        c = __builtin_amdgcn_wmma_f32_16x16x32_f16(false, a.v, false, b.v, (short)0, c,
                                                   false, false);
    }
    // C/D: lane L holds N=L&15; VGPR r holds M = r + 8*(L>>4)
    if (m < DH) {
#pragma unroll
        for (int r = 0; r < 8; ++r) {
            int row = row0 + r + 8 * half;
            if (row < rows) P[row * DH + m] = c[r];
        }
    }
}

// ---------------------------------------------------------------------------
// K2: per-row attention scalars sx, s2 (nodes) and se, s1 (hyperedges)
// ---------------------------------------------------------------------------
__global__ void scalars_kernel(const float* __restrict__ WX, const float* __restrict__ WE,
                               const float* __restrict__ ax, const float* __restrict__ ae,
                               float* __restrict__ sx, float* __restrict__ se,
                               float* __restrict__ s1, float* __restrict__ s2) {
    int i = blockIdx.x * blockDim.x + threadIdx.x;
    if (i < N_NODES) {
        float a = 0.0f, b = 0.0f;
#pragma unroll
        for (int c = 0; c < DH; ++c) {
            float v = WX[i * DH + c];
            a += v * ax[c];        // alpha_x[:dh]
            b += v * ae[DH + c];   // alpha_e[dh:]
        }
        sx[i] = a; s2[i] = b;
    }
    if (i < M_EDGES) {
        float a = 0.0f, b = 0.0f;
#pragma unroll
        for (int c = 0; c < DH; ++c) {
            float v = WE[i * DH + c];
            a += v * ax[DH + c];   // alpha_x[dh:]
            b += v * ae[c];        // alpha_e[:dh]
        }
        se[i] = a; s1[i] = b;
    }
}

// ---------------------------------------------------------------------------
// K3: node->hyperedge pass. One wave per H row: online masked softmax stats,
// then scatter att*WX into E_acc (atomics) and atomicMax of s2 per column.
// ---------------------------------------------------------------------------
__global__ void node_attn_scatter(const float* __restrict__ H,
                                  const float* __restrict__ sx, const float* __restrict__ se,
                                  const float* __restrict__ s2, const float* __restrict__ WX,
                                  float* __restrict__ E_acc, unsigned* __restrict__ t_col_u,
                                  float* __restrict__ U_e) {
    const int lane = threadIdx.x & 31;
    const int wave = threadIdx.x >> 5;
    const int row  = blockIdx.x * (blockDim.x >> 5) + wave;
    if (row >= N_NODES) return;
    const float4* Hr = (const float4*)(H + (size_t)row * M_EDGES);
    const float sxi = sx[row], s2i = s2[row];

    // scan 1: online softmax (max m, sum s) over masked-in entries
    float m = -3.0e38f, s = 0.0f;
#pragma unroll
    for (int j = 0; j < 16; ++j) {
        if (j < 14) __builtin_prefetch(&Hr[lane + 32 * (j + 2)], 0, 0);
        float4 h = Hr[lane + 32 * j];
        float hv[4] = {h.x, h.y, h.z, h.w};
#pragma unroll
        for (int c = 0; c < 4; ++c)
            if (hv[c] > 0.0f) {
                int k = 4 * (lane + 32 * j) + c;
                float l = leaky(sxi + se[k]);
                if (l > m) { s = s * __expf(m - l) + 1.0f; m = l; }
                else       { s += __expf(l - m); }
            }
    }
    for (int off = 16; off; off >>= 1) {
        float om = __shfl_xor(m, off);
        float os = __shfl_xor(s, off);
        float nm = fmaxf(m, om);
        s = s * __expf(m - nm) + os * __expf(om - nm);
        m = nm;
    }

    float wx[DH];
#pragma unroll
    for (int c = 0; c < DH; ++c) wx[c] = WX[row * DH + c];

    if (s > 0.0f) {
        float inv = 1.0f / s;
        unsigned enc = f2o(s2i);
        // scan 2 (L2-resident): scatter
#pragma unroll
        for (int j = 0; j < 16; ++j) {
            float4 h = Hr[lane + 32 * j];
            float hv[4] = {h.x, h.y, h.z, h.w};
#pragma unroll
            for (int c = 0; c < 4; ++c)
                if (hv[c] > 0.0f) {
                    int k = 4 * (lane + 32 * j) + c;
                    float att = __expf(leaky(sxi + se[k]) - m) * inv;
#pragma unroll
                    for (int c2 = 0; c2 < DH; ++c2)
                        atomicAdd(&E_acc[k * DH + c2], att * wx[c2]);
                    atomicMax(&t_col_u[k], enc);
                }
        }
    } else if (lane == 0) {
        // empty row: softmax is uniform 1/M over every hyperedge
#pragma unroll
        for (int c = 0; c < DH; ++c)
            atomicAdd(&U_e[c], wx[c] * (1.0f / (float)M_EDGES));
    }
}

// ---------------------------------------------------------------------------
// K4: E_new = elu(E_acc + U_e); column max m_col = leaky(s1 + max s2); empty
// columns feed the uniform 1/N accumulator U_x.
// ---------------------------------------------------------------------------
__global__ void finalize_e(const float* __restrict__ E_acc, const float* __restrict__ U_e,
                           const unsigned* __restrict__ t_col_u, const float* __restrict__ s1,
                           float* __restrict__ E_new, float* __restrict__ m_col,
                           float* __restrict__ U_x) {
    __shared__ float ec[256];
    int idx = blockIdx.x * 256 + threadIdx.x;  // idx < M_EDGES*DH
    int k = idx >> 3, c = idx & 7;
    float e = elu1(E_acc[idx] + U_e[c]);
    E_new[idx] = e;
    ec[threadIdx.x] = e;
    __syncthreads();
    if (c == 0) {
        unsigned tu = t_col_u[k];
        if (tu == 0u) {  // empty hyperedge: uniform 1/N over all nodes
            m_col[k] = 0.0f;
#pragma unroll
            for (int c2 = 0; c2 < DH; ++c2)
                atomicAdd(&U_x[c2], ec[threadIdx.x + c2] * (1.0f / (float)N_NODES));
        } else {
            m_col[k] = leaky(s1[k] + o2f(tu));  // == max over column (leaky monotone)
        }
    }
}

// ---------------------------------------------------------------------------
// K5: column softmax denominators (row-major scan over H, atomicAdd per nnz)
// ---------------------------------------------------------------------------
__global__ void col_softmax_sum(const float* __restrict__ H, const float* __restrict__ s1,
                                const float* __restrict__ s2, const float* __restrict__ m_col,
                                float* __restrict__ s_col) {
    const int lane = threadIdx.x & 31;
    const int wave = threadIdx.x >> 5;
    const int row  = blockIdx.x * (blockDim.x >> 5) + wave;
    if (row >= N_NODES) return;
    const float4* Hr = (const float4*)(H + (size_t)row * M_EDGES);
    const float s2i = s2[row];
#pragma unroll
    for (int j = 0; j < 16; ++j) {
        if (j < 14) __builtin_prefetch(&Hr[lane + 32 * (j + 2)], 0, 0);
        float4 h = Hr[lane + 32 * j];
        float hv[4] = {h.x, h.y, h.z, h.w};
#pragma unroll
        for (int c = 0; c < 4; ++c)
            if (hv[c] > 0.0f) {
                int k = 4 * (lane + 32 * j) + c;
                atomicAdd(&s_col[k], __expf(leaky(s1[k] + s2i) - m_col[k]));
            }
    }
}

// ---------------------------------------------------------------------------
// K6: hyperedge->node gather: X_new[i] = elu( sum_k att_e[k,i]*E_new[k] + U_x )
// ---------------------------------------------------------------------------
__global__ void edge_to_node(const float* __restrict__ H, const float* __restrict__ s1,
                             const float* __restrict__ s2, const float* __restrict__ m_col,
                             const float* __restrict__ s_col, const float* __restrict__ E_new,
                             const float* __restrict__ U_x, float* __restrict__ X_new) {
    const int lane = threadIdx.x & 31;
    const int wave = threadIdx.x >> 5;
    const int row  = blockIdx.x * (blockDim.x >> 5) + wave;
    if (row >= N_NODES) return;
    const float4* Hr = (const float4*)(H + (size_t)row * M_EDGES);
    const float s2i = s2[row];
    float acc[DH];
#pragma unroll
    for (int c = 0; c < DH; ++c) acc[c] = 0.0f;
#pragma unroll
    for (int j = 0; j < 16; ++j) {
        if (j < 14) __builtin_prefetch(&Hr[lane + 32 * (j + 2)], 0, 0);
        float4 h = Hr[lane + 32 * j];
        float hv[4] = {h.x, h.y, h.z, h.w};
#pragma unroll
        for (int c = 0; c < 4; ++c)
            if (hv[c] > 0.0f) {
                int k = 4 * (lane + 32 * j) + c;
                float w = __expf(leaky(s1[k] + s2i) - m_col[k]) / s_col[k];
#pragma unroll
                for (int c2 = 0; c2 < DH; ++c2) acc[c2] += w * E_new[k * DH + c2];
            }
    }
#pragma unroll
    for (int c = 0; c < DH; ++c)
        for (int off = 16; off; off >>= 1) acc[c] += __shfl_xor(acc[c], off);
    if (lane == 0) {
#pragma unroll
        for (int c = 0; c < DH; ++c) X_new[row * DH + c] = elu1(acc[c] + U_x[c]);
    }
}

// ---------------------------------------------------------------------------
extern "C" void kernel_launch(void* const* d_in, const int* in_sizes, int n_in,
                              void* d_out, int out_size, void* d_ws, size_t ws_size,
                              hipStream_t stream) {
    const float* X  = (const float*)d_in[0];
    const float* E  = (const float*)d_in[1];
    const float* H  = (const float*)d_in[2];
    const float* W  = (const float*)d_in[3];
    const float* ax = (const float*)d_in[4];
    const float* ae = (const float*)d_in[5];

    float* ws = (float*)d_ws;
    float*    WX      = ws;                          // N*8
    float*    WE      = WX + N_NODES * DH;           // M*8
    float*    sx      = WE + M_EDGES * DH;           // N
    float*    se      = sx + N_NODES;                // M
    float*    s1      = se + M_EDGES;                // M
    float*    s2      = s1 + M_EDGES;                // N
    float*    E_acc   = s2 + N_NODES;                // M*8
    float*    E_new   = E_acc + M_EDGES * DH;        // M*8
    unsigned* t_col_u = (unsigned*)(E_new + M_EDGES * DH); // M
    float*    m_col   = (float*)(t_col_u + M_EDGES); // M
    float*    s_col   = m_col + M_EDGES;             // M
    float*    U_e     = s_col + M_EDGES;             // 8
    float*    U_x     = U_e + DH;                    // 8
    float*    X_new   = (float*)d_out;

    init_kernel<<<(M_EDGES * DH + 255) / 256, 256, 0, stream>>>(E_acc, t_col_u, s_col, U_e, U_x);
    proj_wmma<<<N_NODES / 16, 32, 0, stream>>>(X, W, WX, N_NODES);
    proj_wmma<<<M_EDGES / 16, 32, 0, stream>>>(E, W, WE, M_EDGES);
    scalars_kernel<<<(N_NODES + 255) / 256, 256, 0, stream>>>(WX, WE, ax, ae, sx, se, s1, s2);
    node_attn_scatter<<<N_NODES / 8, 256, 0, stream>>>(H, sx, se, s2, WX, E_acc, t_col_u, U_e);
    finalize_e<<<(M_EDGES * DH) / 256, 256, 0, stream>>>(E_acc, U_e, t_col_u, s1, E_new, m_col, U_x);
    col_softmax_sum<<<N_NODES / 8, 256, 0, stream>>>(H, s1, s2, m_col, s_col);
    edge_to_node<<<N_NODES / 8, 256, 0, stream>>>(H, s1, s2, m_col, s_col, E_new, U_x, X_new);
}
